// DictLearnt_IHT_31327491457397
// MI455X (gfx1250) — matrix operand order repaired and checked by reading, not compiled, not run on hardware
//
#include <hip/hip_runtime.h>
#include <hip/hip_bf16.h>
#include <stdint.h>

// ---------------- problem constants ----------------
#define NN    784          // signal dim (rows of W)
#define MM    2048         // dictionary atoms (cols of W)
#define BB    4096         // batch
#define PM_ITER  100
#define IHT_ITER 50
#define KPAD1 800          // K for GEMM1 (784 padded to mult of 32)
#define NPAD2 896          // N tiling pad for GEMM2 (784 -> 7*128)
#define PITCH1 (KPAD1/2)   // 400 u32 pairs per row (K = 800)
#define PITCH2 (MM/2)      // 1024 u32 pairs per row (K = 2048)

// ---------------- vector types (native ext vectors, trivially copyable) ---
typedef __bf16   bf16_t;
typedef bf16_t   v16bf __attribute__((ext_vector_type(16)));
typedef float    v8f   __attribute__((ext_vector_type(8)));
typedef unsigned u32x4 __attribute__((ext_vector_type(4)));
typedef float    f32x4 __attribute__((ext_vector_type(4)));

union FragCast { unsigned u[8]; v16bf v; };

// RNE f32 -> bf16, packed pair (lo half = lower-K element)
__device__ __forceinline__ unsigned pack_bf16(float a, float b) {
  unsigned ua = __float_as_uint(a); ua += 0x7FFFu + ((ua >> 16) & 1u);
  unsigned ub = __float_as_uint(b); ub += 0x7FFFu + ((ub >> 16) & 1u);
  return (ua >> 16) | (ub & 0xFFFF0000u);
}

__device__ __forceinline__ float block_reduce_sum(float v, float* sbuf) {
  int tid = threadIdx.x;
  sbuf[tid] = v; __syncthreads();
  for (int s = 128; s > 0; s >>= 1) {
    if (tid < s) sbuf[tid] += sbuf[tid + s];
    __syncthreads();
  }
  return sbuf[0];
}

// ============================================================
// WMMA GEMM:  C = epilogue(A * B)   with BOTH operands pre-packed
// bf16 pairs:  Xpk[row][pitch] u32, u32 = (bf16 k, bf16 k+1), K = KT*32.
// MODE 0: C = eta * acc
// MODE 1: C = Cprev - eta * acc
// MODE 2: C = acc - Ysub   (cols >= Nstore skipped)
// Block tile 128x128, BK=32, 8 waves (wave32): 4(M) x 2(N),
// each wave 32x64 = 2x4 tiles of v_wmma_f32_16x16x32_bf16.
// Double-buffered LDS, register prefetch, ONE barrier per K-step,
// zero conversion VALU in the hot loop.
// ============================================================
#define BM 128
#define BN 128
#define BK 32
#define KP 16   // BK/2 u32 pairs

template <int MODE>
__global__ __launch_bounds__(256) void gemm_kernel(
    const unsigned* __restrict__ Apk, int pitchA,
    const unsigned* __restrict__ Bpk, int pitchB,
    float* __restrict__ C, int ldc, int Nstore,
    const float* __restrict__ Cprev,
    const float* __restrict__ Ysub, int ldy,
    const float* __restrict__ etaPtr,
    int KT)
{
  __shared__ alignas(16) unsigned As[2][BM * KP];  // 2 x 8 KB
  __shared__ alignas(16) unsigned Bs[2][BN * KP];  // 2 x 8 KB

  const int tid  = threadIdx.x;
  const int bm   = blockIdx.y * BM;
  const int bn   = blockIdx.x * BN;
  const int wave = tid >> 5;
  const int lane = tid & 31;
  const int hi   = lane >> 4;     // lane half selects K sub-range
  const int lrow = lane & 15;
  const int wm   = wave >> 1;     // 0..3 -> 32-row stripe
  const int wn   = wave & 1;      // 0..1 -> 64-col stripe

  v8f acc[2][4];
  const v8f vzero = {0.f,0.f,0.f,0.f,0.f,0.f,0.f,0.f};
#pragma unroll
  for (int mt = 0; mt < 2; ++mt)
#pragma unroll
    for (int nt = 0; nt < 4; ++nt) acc[mt][nt] = vzero;

  // unified loader: 128 rows x 16 u32 per tile, 2 passes of 64 rows
  const int rowb = tid >> 2, c4 = tid & 3;
  const unsigned* Ab = Apk + (size_t)(bm + rowb) * pitchA + c4 * 4;
  const unsigned* Bb = Bpk + (size_t)(bn + rowb) * pitchB + c4 * 4;

  u32x4 aReg[2], bReg[2];

#define LOAD_TILES(ktI) do {                                                   \
    _Pragma("unroll")                                                          \
    for (int p = 0; p < 2; ++p) {                                              \
      aReg[p] = *reinterpret_cast<const u32x4*>(Ab + (size_t)p*64*pitchA + (ktI)*KP); \
      bReg[p] = *reinterpret_cast<const u32x4*>(Bb + (size_t)p*64*pitchB + (ktI)*KP); \
    }                                                                          \
  } while (0)

#define STORE_TILES(buf) do {                                                  \
    _Pragma("unroll")                                                          \
    for (int p = 0; p < 2; ++p) {                                              \
      *reinterpret_cast<u32x4*>(&As[buf][(rowb + p*64) * KP + c4*4]) = aReg[p]; \
      *reinterpret_cast<u32x4*>(&Bs[buf][(rowb + p*64) * KP + c4*4]) = bReg[p]; \
    }                                                                          \
  } while (0)

  LOAD_TILES(0);
  STORE_TILES(0);
  __syncthreads();

  for (int kt = 0; kt < KT; ++kt) {
    const int  cur  = kt & 1;
    const bool more = (kt + 1) < KT;
    if (more) LOAD_TILES(kt + 1);   // global prefetch overlapped with WMMA

    // ---- fragments per ISA 7.12.2 16-bit layouts ----
    v16bf afrag[2], bfrag[4];
#pragma unroll
    for (int mt = 0; mt < 2; ++mt) {
      int arow = wm * 32 + mt * 16 + lrow;
      FragCast fc;
      *reinterpret_cast<u32x4*>(&fc.u[0]) =
          *reinterpret_cast<const u32x4*>(&As[cur][arow * KP + hi * 4]);      // K 0..7 | 8..15
      *reinterpret_cast<u32x4*>(&fc.u[4]) =
          *reinterpret_cast<const u32x4*>(&As[cur][arow * KP + 8 + hi * 4]);  // K 16..23 | 24..31
      afrag[mt] = fc.v;
    }
#pragma unroll
    for (int nt = 0; nt < 4; ++nt) {
      int brow = wn * 64 + nt * 16 + lrow;
      FragCast fc;
      *reinterpret_cast<u32x4*>(&fc.u[0]) =
          *reinterpret_cast<const u32x4*>(&Bs[cur][brow * KP + hi * 4]);      // K 0..7 | 8..15
      *reinterpret_cast<u32x4*>(&fc.u[4]) =
          *reinterpret_cast<const u32x4*>(&Bs[cur][brow * KP + 8 + hi * 4]);  // K 16..23 | 24..31
      bfrag[nt] = fc.v;
    }
#pragma unroll
    for (int mt = 0; mt < 2; ++mt)
#pragma unroll
      for (int nt = 0; nt < 4; ++nt)
        acc[mt][nt] = __builtin_amdgcn_wmma_f32_16x16x32_bf16(
            false, afrag[mt], false, bfrag[nt], (short)0, acc[mt][nt], false, false);

    if (more) {
      STORE_TILES(cur ^ 1);
      __syncthreads();   // single barrier per K-step (ping-pong buffers)
    }
  }
#undef LOAD_TILES
#undef STORE_TILES

  float eta = (MODE == 2) ? 1.0f : etaPtr[0];
#pragma unroll
  for (int mt = 0; mt < 2; ++mt) {
#pragma unroll
    for (int nt = 0; nt < 4; ++nt) {
      int gn = bn + wn * 64 + nt * 16 + lrow;
      if (gn >= Nstore) continue;
#pragma unroll
      for (int e = 0; e < 8; ++e) {
        int gm = bm + wm * 32 + mt * 16 + hi * 8 + e;  // C/D: lanes 16-31 hold M=8..15
        size_t off = (size_t)gm * ldc + gn;
        float val = acc[mt][nt][e];
        if (MODE == 0)      C[off] = eta * val;
        else if (MODE == 1) C[off] = Cprev[off] - eta * val;
        else                C[off] = val - Ysub[(size_t)gm * ldy + gn];
      }
    }
  }
}

// ============================================================
// Row-wise hard threshold: keep x where |x| > (K+1)-th largest |x|.
// One 256-thread block per row of 2048, thread t owns contiguous
// cols [8t, 8t+8). 4-pass radix select on the f32 abs bit pattern.
// Emits Gamma f32 AND Gamma packed bf16 pairs (WMMA A operand).
// ============================================================
__global__ __launch_bounds__(256) void topk_threshold_kernel(
    const float* __restrict__ U, float* __restrict__ G,
    unsigned* __restrict__ Gpk, const int* __restrict__ Kptr)
{
  const int row = blockIdx.x;
  const int tid = threadIdx.x;
  const float* src = U + (size_t)row * MM + 8 * tid;
  f32x4 va = *reinterpret_cast<const f32x4*>(src);
  f32x4 vb = *reinterpret_cast<const f32x4*>(src + 4);
  float vals[8] = {va.x, va.y, va.z, va.w, vb.x, vb.y, vb.z, vb.w};
  unsigned ab[8];
#pragma unroll
  for (int i = 0; i < 8; ++i) ab[i] = __float_as_uint(vals[i]) & 0x7FFFFFFFu;

  __shared__ unsigned hist[256];
  __shared__ unsigned s_prefix, s_rank;
  if (tid == 0) { s_prefix = 0u; s_rank = (unsigned)(Kptr[0] + 1); }
  __syncthreads();

  for (int pass = 0; pass < 4; ++pass) {
    const int shift = 24 - 8 * pass;
    hist[tid] = 0u;
    __syncthreads();
    unsigned pfx = s_prefix;
#pragma unroll
    for (int i = 0; i < 8; ++i) {
      unsigned a = ab[i];
      bool match = (pass == 0) || ((a >> (shift + 8)) == pfx);
      if (match) atomicAdd(&hist[(a >> shift) & 255u], 1u);
    }
    __syncthreads();
    if (tid == 0) {
      unsigned r = s_rank, cum = 0u; int sel = 0;
      for (int b = 255; b >= 0; --b) {
        unsigned h = hist[b];
        if (cum + h >= r) { sel = b; s_rank = r - cum; break; }
        cum += h;
      }
      s_prefix = (s_prefix << 8) | (unsigned)sel;
    }
    __syncthreads();
  }
  const unsigned Tbits = s_prefix;   // bit pattern of the (K+1)-th largest |x|

  float o[8];
#pragma unroll
  for (int i = 0; i < 8; ++i) o[i] = (ab[i] > Tbits) ? vals[i] : 0.0f;

  float* dst = G + (size_t)row * MM + 8 * tid;
  f32x4 oa = {o[0], o[1], o[2], o[3]};
  f32x4 ob = {o[4], o[5], o[6], o[7]};
  *reinterpret_cast<f32x4*>(dst)     = oa;
  *reinterpret_cast<f32x4*>(dst + 4) = ob;

  u32x4 pk = {pack_bf16(o[0], o[1]), pack_bf16(o[2], o[3]),
              pack_bf16(o[4], o[5]), pack_bf16(o[6], o[7])};
  *reinterpret_cast<u32x4*>(Gpk + (size_t)row * PITCH2 + 4 * tid) = pk;
}

// ============================================================
// Prep kernels
// ============================================================
__global__ void colnorm_kernel(const float* __restrict__ W, float* __restrict__ norms) {
  int j = blockIdx.x * 256 + threadIdx.x;
  if (j >= MM) return;
  float s = 0.f;
  for (int i = 0; i < NN; ++i) { float v = W[(size_t)i * MM + j]; s += v * v; }
  norms[j] = sqrtf(s);
}

__global__ void wn_kernel(const float* __restrict__ W, const float* __restrict__ norms,
                          float* __restrict__ Wn) {
  int idx = blockIdx.x * 256 + threadIdx.x;
  if (idx >= NN * MM) return;
  Wn[idx] = W[idx] / norms[idx & (MM - 1)];
}

// Bpk1[j][p] = bf16 pair (Wn[2p][j], Wn[2p+1][j]);  [MM][PITCH1], K pad zeroed
__global__ void packB1_kernel(const float* __restrict__ Wn, unsigned* __restrict__ Bpk1) {
  int idx = blockIdx.x * 256 + threadIdx.x;
  if (idx >= MM * PITCH1) return;
  int j = idx / PITCH1, p = idx % PITCH1;
  int i0 = 2 * p, i1 = 2 * p + 1;
  float a = (i0 < NN) ? Wn[(size_t)i0 * MM + j] : 0.f;
  float b = (i1 < NN) ? Wn[(size_t)i1 * MM + j] : 0.f;
  Bpk1[idx] = pack_bf16(a, b);
}

// Bpk2[n][p] = bf16 pair (Wn[n][2p], Wn[n][2p+1]);  [NPAD2][PITCH2], rows >=784 zero
__global__ void packB2_kernel(const float* __restrict__ Wn, unsigned* __restrict__ Bpk2) {
  int idx = blockIdx.x * 256 + threadIdx.x;
  if (idx >= NPAD2 * PITCH2) return;
  int n = idx >> 10, p = idx & (PITCH2 - 1);
  float a = 0.f, b = 0.f;
  if (n < NN) { a = Wn[(size_t)n * MM + 2 * p]; b = Wn[(size_t)n * MM + 2 * p + 1]; }
  Bpk2[idx] = pack_bf16(a, b);
}

// Generic row packer: dst[row][p] = bf16pair(src[row][2p], src[row][2p+1]), pad->0
__global__ void pack_rows_kernel(const float* __restrict__ src, int srcPitch, int srcCols,
                                 unsigned* __restrict__ dst, int dstPitch) {
  int p   = blockIdx.x * 256 + threadIdx.x;
  int row = blockIdx.y;
  if (p >= dstPitch) return;
  const float* s = src + (size_t)row * srcPitch;
  int c0 = 2 * p, c1 = 2 * p + 1;
  float a = (c0 < srcCols) ? s[c0] : 0.f;
  float b = (c1 < srcCols) ? s[c1] : 0.f;
  dst[(size_t)row * dstPitch + p] = pack_bf16(a, b);
}

// ============================================================
// Power method (f32, small)
// ============================================================
__global__ void pm_rowdot_kernel(const float* __restrict__ Wn, const float* __restrict__ x,
                                 float* __restrict__ v) {
  __shared__ float sbuf[256];
  const float* row = Wn + (size_t)blockIdx.x * MM;
  float s = 0.f;
  for (int t = threadIdx.x; t < MM; t += 256) s += row[t] * x[t];
  s = block_reduce_sum(s, sbuf);
  if (threadIdx.x == 0) v[blockIdx.x] = s;
}

__global__ void pm_coldot_kernel(const float* __restrict__ Wn, const float* __restrict__ v,
                                 float* __restrict__ xnew) {
  int j = blockIdx.x * 256 + threadIdx.x;
  if (j >= MM) return;
  float s = 0.f;
  for (int i = 0; i < NN; ++i) s += v[i] * Wn[(size_t)i * MM + j];
  xnew[j] = s;
}

__global__ void pm_normalize_kernel(float* __restrict__ x, const float* __restrict__ xnew,
                                    float* __restrict__ cOut) {
  __shared__ float sbuf[256];
  float s = 0.f;
  for (int t = threadIdx.x; t < MM; t += 256) { float u = xnew[t]; s += u * u; }
  s = block_reduce_sum(s, sbuf);
  float nm = sqrtf(s);
  for (int t = threadIdx.x; t < MM; t += 256) x[t] = xnew[t] / nm;
  if (threadIdx.x == 0) cOut[0] = nm;
}

__global__ void eta_kernel(const float* __restrict__ c, float* __restrict__ eta) {
  if (threadIdx.x == 0) eta[0] = 2.0f / c[0];
}

// ============================================================
// Deterministic two-stage sum-of-squares reduction
// ============================================================
__global__ void sumsq_partial_kernel(const float* __restrict__ X, size_t n,
                                     float* __restrict__ partials) {
  __shared__ float sbuf[256];
  size_t stride = (size_t)gridDim.x * 256;
  float s = 0.f;
  for (size_t t = (size_t)blockIdx.x * 256 + threadIdx.x; t < n; t += stride) {
    float v = X[t]; s += v * v;
  }
  s = block_reduce_sum(s, sbuf);
  if (threadIdx.x == 0) partials[blockIdx.x] = s;
}

__global__ void norm_final_kernel(const float* __restrict__ partials, int np,
                                  const float* __restrict__ divPtr, float* __restrict__ out) {
  __shared__ float sbuf[256];
  float s = (threadIdx.x < np) ? partials[threadIdx.x] : 0.f;
  s = block_reduce_sum(s, sbuf);
  if (threadIdx.x == 0) {
    float r = sqrtf(s);
    out[0] = divPtr ? (r / divPtr[0]) : r;
  }
}

// X = residual + Y   (all pitch 784)
__global__ void add_kernel(const float* __restrict__ resid, const float* __restrict__ Y,
                           float* __restrict__ X, size_t n) {
  size_t i = (size_t)blockIdx.x * 256 + threadIdx.x;
  if (i < n) X[i] = resid[i] + Y[i];
}

// ============================================================
// Workspace layout (bytes, all >=256B aligned)
// ============================================================
#define WN_OFF     ((size_t)0)
#define BPK1_OFF   (WN_OFF    + (size_t)NN * MM * 4)          //  6,422,528
#define BPK2_OFF   (BPK1_OFF  + (size_t)MM * PITCH1 * 4)      // +3,276,800
#define U_OFF      (BPK2_OFF  + (size_t)NPAD2 * PITCH2 * 4)   // +3,670,016
#define RES_OFF    (U_OFF     + (size_t)BB * MM * 4)          // +33,554,432
#define YPK_OFF    (RES_OFF   + (size_t)BB * NN * 4)          // +12,845,056
#define RESPK_OFF  (YPK_OFF   + (size_t)BB * PITCH1 * 4)      // +6,553,600
#define GPK_OFF    (RESPK_OFF + (size_t)BB * PITCH1 * 4)      // +6,553,600
#define NORMS_OFF  (GPK_OFF   + (size_t)BB * PITCH2 * 4)      // +16,777,216
#define XV_OFF     (NORMS_OFF + 8192)
#define XNEW_OFF   (XV_OFF    + 8192)
#define VV_OFF     (XNEW_OFF  + 8192)
#define PART_OFF   (VV_OFF    + 4096)
#define SCAL_OFF   (PART_OFF  + 1024)

extern "C" void kernel_launch(void* const* d_in, const int* in_sizes, int n_in,
                              void* d_out, int out_size, void* d_ws, size_t ws_size,
                              hipStream_t stream) {
  (void)in_sizes; (void)n_in; (void)out_size; (void)ws_size;
  const float* Y  = (const float*)d_in[0];
  const float* W  = (const float*)d_in[1];
  const float* pm = (const float*)d_in[2];
  const int*   Kp = (const int*)d_in[3];

  char* ws = (char*)d_ws;
  float*    Wn     = (float*)(ws + WN_OFF);
  unsigned* Bpk1   = (unsigned*)(ws + BPK1_OFF);
  unsigned* Bpk2   = (unsigned*)(ws + BPK2_OFF);
  float*    U      = (float*)(ws + U_OFF);
  float*    resid  = (float*)(ws + RES_OFF);
  unsigned* Ypk    = (unsigned*)(ws + YPK_OFF);
  unsigned* residPk= (unsigned*)(ws + RESPK_OFF);
  unsigned* Gpk    = (unsigned*)(ws + GPK_OFF);
  float*    norms  = (float*)(ws + NORMS_OFF);
  float*    xv     = (float*)(ws + XV_OFF);
  float*    xnew   = (float*)(ws + XNEW_OFF);
  float*    vv     = (float*)(ws + VV_OFF);
  float*    part   = (float*)(ws + PART_OFF);
  float*    scal   = (float*)(ws + SCAL_OFF);   // [0]=c, [1]=eta, [2]=||Y||

  float* outX = (float*)d_out;
  float* outG = outX + (size_t)BB * NN;
  float* outE = outG + (size_t)BB * MM;

  // ---- prep: normalize W, pack all static bf16 operands ----
  colnorm_kernel<<<MM / 256, 256, 0, stream>>>(W, norms);
  wn_kernel<<<(NN * MM) / 256, 256, 0, stream>>>(W, norms, Wn);
  packB1_kernel<<<(MM * PITCH1) / 256, 256, 0, stream>>>(Wn, Bpk1);
  packB2_kernel<<<(NPAD2 * PITCH2) / 256, 256, 0, stream>>>(Wn, Bpk2);
  pack_rows_kernel<<<dim3(2, BB), 256, 0, stream>>>(Y, NN, NN, Ypk, PITCH1);

  // ---- ||Y|| ----
  sumsq_partial_kernel<<<256, 256, 0, stream>>>(Y, (size_t)BB * NN, part);
  norm_final_kernel<<<1, 256, 0, stream>>>(part, 256, nullptr, scal + 2);

  // ---- power method: c = spectral-norm estimate, eta = 2/c ----
  hipMemcpyAsync(xv, pm, MM * sizeof(float), hipMemcpyDeviceToDevice, stream);
  for (int it = 0; it < PM_ITER; ++it) {
    pm_rowdot_kernel<<<NN, 256, 0, stream>>>(Wn, xv, vv);
    pm_coldot_kernel<<<MM / 256, 256, 0, stream>>>(Wn, vv, xnew);
    pm_normalize_kernel<<<1, 256, 0, stream>>>(xv, xnew, scal + 0);
  }
  eta_kernel<<<1, 64, 0, stream>>>(scal + 0, scal + 1);

  const dim3 g1(MM / BN, BB / BM);      // (16, 32): out 4096 x 2048
  const dim3 g2(NPAD2 / BN, BB / BM);   // ( 7, 32): out 4096 x 784

  // ---- Gamma0 = HT(Y @ (eta*Wn));  residual = Gamma0 @ Wn^T - Y ----
  gemm_kernel<0><<<g1, 256, 0, stream>>>(Ypk, PITCH1, Bpk1, PITCH1, U, MM, MM,
                                         nullptr, nullptr, 0, scal + 1, KPAD1 / BK);
  topk_threshold_kernel<<<BB, 256, 0, stream>>>(U, outG, Gpk, Kp);
  gemm_kernel<2><<<g2, 256, 0, stream>>>(Gpk, PITCH2, Bpk2, PITCH2, resid, NN, NN,
                                         nullptr, Y, NN, scal + 1, MM / BK);

  // ---- 50 IHT iterations ----
  for (int it = 0; it < IHT_ITER; ++it) {
    pack_rows_kernel<<<dim3(2, BB), 256, 0, stream>>>(resid, NN, NN, residPk, PITCH1);
    gemm_kernel<1><<<g1, 256, 0, stream>>>(residPk, PITCH1, Bpk1, PITCH1, U, MM, MM,
                                           outG, nullptr, 0, scal + 1, KPAD1 / BK);
    topk_threshold_kernel<<<BB, 256, 0, stream>>>(U, outG, Gpk, Kp);
    gemm_kernel<2><<<g2, 256, 0, stream>>>(Gpk, PITCH2, Bpk2, PITCH2, resid, NN, NN,
                                           nullptr, Y, NN, scal + 1, MM / BK);
    sumsq_partial_kernel<<<256, 256, 0, stream>>>(resid, (size_t)BB * NN, part);
    norm_final_kernel<<<1, 256, 0, stream>>>(part, 256, scal + 2, outE + it);
  }

  // ---- X = Gamma @ Wn^T = residual + Y ----
  add_kernel<<<((size_t)BB * NN) / 256, 256, 0, stream>>>(resid, Y, outX, (size_t)BB * NN);
}
